// NVAN_75892072120467
// MI455X (gfx1250) — matrix operand order repaired and checked by reading, not compiled, not run on hardware
//
#include <hip/hip_runtime.h>
#include <hip/hip_bf16.h>

typedef unsigned short u16;
typedef unsigned int   u32;
typedef u32   u32x4 __attribute__((ext_vector_type(4)));
typedef __bf16 bf16_t;
typedef bf16_t v16bf __attribute__((ext_vector_type(16)));
typedef float  v8f   __attribute__((ext_vector_type(8)));

union BFrag { v16bf v; u32x4 q[2]; };

__device__ __forceinline__ u16 f2bf(float f) {
    u32 u = __float_as_uint(f);
    u += 0x7fffu + ((u >> 16) & 1u);
    return (u16)(u >> 16);
}
__device__ __forceinline__ float sigm(float x) { return 1.0f / (1.0f + expf(-x)); }

// ---------------------------------------------------------------------------
// Weight conversion f32 -> bf16
// ---------------------------------------------------------------------------
__global__ void cvt_bf16_kernel(const float* __restrict__ src, u16* __restrict__ dst, int n) {
    int i = blockIdx.x * 256 + threadIdx.x;
    if (i < n) dst[i] = f2bf(src[i]);
}

// ---------------------------------------------------------------------------
// LSTM layer 0 (in_dim = 1).  One block per (view, dir). 8 waves.
// Per step: recurrent GEMM (128x256)@(256x1024) via bf16 WMMA, f32 accum.
// H double-buffered bf16 in LDS; cell state c in f32 LDS (each lane owns its
// own slots); M-tiles processed in pairs so each B fragment feeds 2 WMMAs and
// the live set stays ~190 VGPRs (no spills, no VGPR-MSB juggling).
// out1 layout: (v, 32, 128, 512) bf16  [fwd cols 0..255, bwd cols 256..511]
// ---------------------------------------------------------------------------
__global__ __launch_bounds__(256) void lstm_l0_kernel(
    const float* __restrict__ X,                    // (32,128,16)
    const float* __restrict__ wih0f, const float* __restrict__ wih0b,  // (1024)
    const u16*   __restrict__ whh,                  // (2,1024,256) bf16
    const float* __restrict__ biasf, const float* __restrict__ biasb,  // (1024)
    u16*         __restrict__ out1)                 // (16,32,128,512) bf16
{
    extern __shared__ char smem[];
    u16*   Hbuf = (u16*)smem;                         // 2 * 128*256 bf16 (128 KB)
    float* Cls  = (float*)(smem + 2 * 128 * 256 * 2); // 128*256 f32    (128 KB)
    float* Xf   = (float*)(smem + 2 * 128 * 256 * 2 + 128 * 256 * 4);  // 128 f32

    const int v = blockIdx.x >> 1, d = blockIdx.x & 1;
    const int wave = threadIdx.x >> 5, lane = threadIdx.x & 31;
    const int laneLo = lane & 15, laneHi = lane >> 4;

    const u16*   Whh  = whh + (size_t)d * 1024 * 256;
    const float* bias = d ? biasb : biasf;
    const float* wih0 = d ? wih0b : wih0f;

    for (int i = threadIdx.x; i < 128 * 256; i += 256) { Hbuf[i] = 0; Cls[i] = 0.f; }
    __syncthreads();

#define GATES_L0(ACC, MT)                                                      \
    {                                                                          \
        _Pragma("unroll")                                                      \
        for (int t = 0; t < 2; ++t) {                                          \
            const int hcol = wave * 32 + t * 16 + laneLo;                      \
            const float bi = bias[hcol], bff = bias[256 + hcol];               \
            const float bg = bias[512 + hcol], bo = bias[768 + hcol];          \
            const float wi = wih0[hcol], wf = wih0[256 + hcol];                \
            const float wg = wih0[512 + hcol], wo = wih0[768 + hcol];          \
            _Pragma("unroll")                                                  \
            for (int r = 0; r < 8; ++r) {                                      \
                const int m = (MT) * 16 + laneHi * 8 + r;                      \
                const float xv = Xf[m];                                        \
                const float zi = ACC[0 + t][r] + bi + xv * wi;                 \
                const float zf = ACC[2 + t][r] + bff + xv * wf;                \
                const float zg = ACC[4 + t][r] + bg + xv * wg;                 \
                const float zo = ACC[6 + t][r] + bo + xv * wo;                 \
                float c = Cls[m * 256 + hcol];                                 \
                c = sigm(zf) * c + sigm(zi) * tanhf(zg);                       \
                Cls[m * 256 + hcol] = c;                                       \
                const float h = sigm(zo) * tanhf(c);                           \
                const u16 hb = f2bf(h);                                        \
                Hw[m * 256 + hcol] = hb;                                       \
                out1[(((size_t)v * 32 + bidx) * 128 + m) * 512 + d * 256 + hcol] = hb; \
            }                                                                  \
        }                                                                      \
    }

#pragma clang loop unroll(disable)
    for (int s = 0; s < 32; ++s) {
        const int bidx = d ? (31 - s) : s;
        u16* Hr = Hbuf + (s & 1) * (128 * 256);
        u16* Hw = Hbuf + ((s + 1) & 1) * (128 * 256);

        if (threadIdx.x < 128)
            Xf[threadIdx.x] = X[(size_t)bidx * 128 * 16 + threadIdx.x * 16 + v];
        __syncthreads();

#pragma unroll
        for (int mg = 0; mg < 4; ++mg) {
            v8f acc0[8], acc1[8];
#pragma unroll
            for (int j = 0; j < 8; ++j) {
                acc0[j] = v8f{0.f,0.f,0.f,0.f,0.f,0.f,0.f,0.f};
                acc1[j] = v8f{0.f,0.f,0.f,0.f,0.f,0.f,0.f,0.f};
            }
            const int arow0 = mg * 32 + laneLo;       // rows of M-tile 2*mg
            const int arow1 = arow0 + 16;             // rows of M-tile 2*mg+1

#pragma clang loop unroll_count(2)
            for (int kki = 0; kki < 8; ++kki) {
                const int kk = kki * 32;
                BFrag A0, A1;
                const u16* ap0 = Hr + arow0 * 256 + kk + laneHi * 8;
                const u16* ap1 = Hr + arow1 * 256 + kk + laneHi * 8;
                A0.q[0] = *(const u32x4*)(ap0);
                A0.q[1] = *(const u32x4*)(ap0 + 16);
                A1.q[0] = *(const u32x4*)(ap1);
                A1.q[1] = *(const u32x4*)(ap1 + 16);
#pragma unroll
                for (int g = 0; g < 4; ++g)
#pragma unroll
                    for (int t = 0; t < 2; ++t) {
                        const int ncol = g * 256 + wave * 32 + t * 16 + laneLo;
                        const u16* bp = Whh + (size_t)ncol * 256 + kk + laneHi * 8;
                        BFrag Bm;
                        Bm.q[0] = *(const u32x4*)(bp);
                        Bm.q[1] = *(const u32x4*)(bp + 16);
                        acc0[g * 2 + t] = __builtin_amdgcn_wmma_f32_16x16x32_bf16(
                            false, A0.v, false, Bm.v, (short)0, acc0[g * 2 + t], false, false);
                        acc1[g * 2 + t] = __builtin_amdgcn_wmma_f32_16x16x32_bf16(
                            false, A1.v, false, Bm.v, (short)0, acc1[g * 2 + t], false, false);
                    }
            }
            GATES_L0(acc0, 2 * mg)
            GATES_L0(acc1, 2 * mg + 1)
        }
        __syncthreads();
    }
#undef GATES_L0
}

// ---------------------------------------------------------------------------
// LSTM layer 1 (in_dim = 512).  Combined K = 768 (input 512 + recurrent 256).
// A-fragments of the input tile are read straight from global (L2-resident,
// prefetched); recurrent A-fragments from LDS H.  c state in f32 LDS.
// hid layout: (32, 16, 128, 512) f32
// ---------------------------------------------------------------------------
__global__ __launch_bounds__(256) void lstm_l1_kernel(
    const u16*   __restrict__ in1,   // (16,32,128,512) bf16
    const u16*   __restrict__ wih,   // (2,1024,512) bf16
    const u16*   __restrict__ whh,   // (2,1024,256) bf16
    const float* __restrict__ biasf, const float* __restrict__ biasb,
    float*       __restrict__ hid)   // (32,16,128,512) f32
{
    extern __shared__ char smem[];
    u16*   Hbuf = (u16*)smem;                         // 2 * 128*256 bf16 (128 KB)
    float* Cls  = (float*)(smem + 2 * 128 * 256 * 2); // 128*256 f32    (128 KB)

    const int v = blockIdx.x >> 1, d = blockIdx.x & 1;
    const int wave = threadIdx.x >> 5, lane = threadIdx.x & 31;
    const int laneLo = lane & 15, laneHi = lane >> 4;

    const u16*   Wih  = wih + (size_t)d * 1024 * 512;
    const u16*   Whh  = whh + (size_t)d * 1024 * 256;
    const float* bias = d ? biasb : biasf;

    for (int i = threadIdx.x; i < 128 * 256; i += 256) { Hbuf[i] = 0; Cls[i] = 0.f; }
    __syncthreads();

#define GATES_L1(ACC, MT)                                                      \
    {                                                                          \
        _Pragma("unroll")                                                      \
        for (int t = 0; t < 2; ++t) {                                          \
            const int hcol = wave * 32 + t * 16 + laneLo;                      \
            const float bi = bias[hcol], bff = bias[256 + hcol];               \
            const float bg = bias[512 + hcol], bo = bias[768 + hcol];          \
            _Pragma("unroll")                                                  \
            for (int r = 0; r < 8; ++r) {                                      \
                const int m = (MT) * 16 + laneHi * 8 + r;                      \
                const float zi = ACC[0 + t][r] + bi;                           \
                const float zf = ACC[2 + t][r] + bff;                          \
                const float zg = ACC[4 + t][r] + bg;                           \
                const float zo = ACC[6 + t][r] + bo;                           \
                float c = Cls[m * 256 + hcol];                                 \
                c = sigm(zf) * c + sigm(zi) * tanhf(zg);                       \
                Cls[m * 256 + hcol] = c;                                       \
                const float h = sigm(zo) * tanhf(c);                           \
                Hw[m * 256 + hcol] = f2bf(h);                                  \
                hid[(((size_t)bidx * 16 + v) * 128 + m) * 512 + d * 256 + hcol] = h; \
            }                                                                  \
        }                                                                      \
    }

#pragma clang loop unroll(disable)
    for (int s = 0; s < 32; ++s) {
        const int bidx = d ? (31 - s) : s;
        u16* Hr = Hbuf + (s & 1) * (128 * 256);
        u16* Hw = Hbuf + ((s + 1) & 1) * (128 * 256);
        const u16* xbase = in1 + ((size_t)v * 32 + bidx) * 128 * 512;

        // warm L2 for next step's input tile (global_prefetch_b8)
        if (s + 1 < 32) {
            const int nb = d ? (31 - (s + 1)) : (s + 1);
            const u16* nxt = in1 + ((size_t)v * 32 + nb) * 128 * 512;
            __builtin_prefetch(nxt + threadIdx.x * 256, 0, 1);
        }

#pragma unroll
        for (int mg = 0; mg < 4; ++mg) {
            v8f acc0[8], acc1[8];
#pragma unroll
            for (int j = 0; j < 8; ++j) {
                acc0[j] = v8f{0.f,0.f,0.f,0.f,0.f,0.f,0.f,0.f};
                acc1[j] = v8f{0.f,0.f,0.f,0.f,0.f,0.f,0.f,0.f};
            }
            const int arow0 = mg * 32 + laneLo;
            const int arow1 = arow0 + 16;

#pragma clang loop unroll_count(2)
            for (int kki = 0; kki < 24; ++kki) {
                const int kk = kki * 32;
                BFrag A0, A1;
                if (kk < 512) {
                    const u16* ap0 = xbase + (size_t)arow0 * 512 + kk + laneHi * 8;
                    const u16* ap1 = xbase + (size_t)arow1 * 512 + kk + laneHi * 8;
                    A0.q[0] = *(const u32x4*)(ap0);
                    A0.q[1] = *(const u32x4*)(ap0 + 16);
                    A1.q[0] = *(const u32x4*)(ap1);
                    A1.q[1] = *(const u32x4*)(ap1 + 16);
                } else {
                    const u16* ap0 = Hr + arow0 * 256 + (kk - 512) + laneHi * 8;
                    const u16* ap1 = Hr + arow1 * 256 + (kk - 512) + laneHi * 8;
                    A0.q[0] = *(const u32x4*)(ap0);
                    A0.q[1] = *(const u32x4*)(ap0 + 16);
                    A1.q[0] = *(const u32x4*)(ap1);
                    A1.q[1] = *(const u32x4*)(ap1 + 16);
                }
#pragma unroll
                for (int g = 0; g < 4; ++g)
#pragma unroll
                    for (int t = 0; t < 2; ++t) {
                        const int ncol = g * 256 + wave * 32 + t * 16 + laneLo;
                        const u16* bp = (kk < 512)
                            ? (Wih + (size_t)ncol * 512 + kk + laneHi * 8)
                            : (Whh + (size_t)ncol * 256 + (kk - 512) + laneHi * 8);
                        BFrag Bm;
                        Bm.q[0] = *(const u32x4*)(bp);
                        Bm.q[1] = *(const u32x4*)(bp + 16);
                        acc0[g * 2 + t] = __builtin_amdgcn_wmma_f32_16x16x32_bf16(
                            false, A0.v, false, Bm.v, (short)0, acc0[g * 2 + t], false, false);
                        acc1[g * 2 + t] = __builtin_amdgcn_wmma_f32_16x16x32_bf16(
                            false, A1.v, false, Bm.v, (short)0, acc1[g * 2 + t], false, false);
                    }
            }
            GATES_L1(acc0, 2 * mg)
            GATES_L1(acc1, 2 * mg + 1)
        }
        __syncthreads();
    }
#undef GATES_L1
}

// ---------------------------------------------------------------------------
// h_t and cc1 (from h_last).  grid = B(32), block = 128 (t')
// ---------------------------------------------------------------------------
__global__ void ht_cc1_kernel(const float* __restrict__ hid,
                              const float* __restrict__ w_sc_t, const float* __restrict__ b_sc_t,
                              const float* __restrict__ w_cc1,  const float* __restrict__ b_cc1,
                              float* __restrict__ h_t, float* __restrict__ cc1)
{
    const int b = blockIdx.x, tp = threadIdx.x;
    float st = 0.f, sc = 0.f;
    for (int v = 0; v < 16; ++v) {
        const float* base = hid + (((size_t)b * 16 + v) * 128 + 127) * 512;
        const float* wt = w_sc_t + v * 385;
        const float* wc = w_cc1 + v * 385;
        for (int k = 0; k < 385; ++k) {
            float hv = base[k + tp];
            st += hv * wt[k];
            sc += hv * wc[k];
        }
    }
    float sb = 0.f;
    for (int v = 0; v < 16; ++v) sb += b_sc_t[v];
    h_t[b * 128 + tp] = st + sb;
    cc1[b * 128 + tp] = sc + b_cc1[0];
}

// ---------------------------------------------------------------------------
// h_i and cc2 (from h_prev).  grid = B*127, block = 128
// ---------------------------------------------------------------------------
__global__ void hi_cc2_kernel(const float* __restrict__ hid,
                              const float* __restrict__ w_sc_i, const float* __restrict__ b_sc_i,
                              const float* __restrict__ w_cc2,  const float* __restrict__ b_cc2,
                              float* __restrict__ h_i, float* __restrict__ cc2)
{
    const int b = blockIdx.x / 127, s = blockIdx.x % 127, tp = threadIdx.x;
    float si = 0.f, sc = 0.f;
    for (int v = 0; v < 16; ++v) {
        const float* base = hid + (((size_t)b * 16 + v) * 128 + s) * 512;
        const float* wi = w_sc_i + v * 385;
        const float* wc = w_cc2 + v * 385;
        for (int k = 0; k < 385; ++k) {
            float hv = base[k + tp];
            si += hv * wi[k];
            sc += hv * wc[k];
        }
    }
    float sb = 0.f;
    for (int v = 0; v < 16; ++v) sb += b_sc_i[v];
    h_i[((size_t)b * 127 + s) * 128 + tp] = si + sb;
    cc2[((size_t)b * 127 + s) * 128 + tp] = sc + b_cc2[0];
}

// ---------------------------------------------------------------------------
// dots[b,v,t] = h_prev[b,v,t,:] . h_last[b,v,:]   grid = B*V, block = 128
// ---------------------------------------------------------------------------
__global__ void dots_kernel(const float* __restrict__ hid, float* __restrict__ dots)
{
    const int bv = blockIdx.x, t = threadIdx.x;
    if (t >= 127) return;
    const float* hp = hid + ((size_t)bv * 128 + t) * 512;
    const float* hl = hid + ((size_t)bv * 128 + 127) * 512;
    float acc = 0.f;
    for (int k = 0; k < 512; ++k) acc += hp[k] * hl[k];
    dots[bv * 127 + t] = acc;
}

// ---------------------------------------------------------------------------
// soft[b,j,v] = softmax_v dots[b,v,prev_idx[j]]   grid = B, block = 128
// ---------------------------------------------------------------------------
__global__ void soft_kernel(const float* __restrict__ dots, float* __restrict__ soft)
{
    const int b = blockIdx.x, j = threadIdx.x;
    if (j >= 127) return;
    const int pidx = (j == 0) ? 0 : (j - 1);
    float vals[16];
    float mx = -1e30f;
    for (int v = 0; v < 16; ++v) {
        vals[v] = dots[(b * 16 + v) * 127 + pidx];
        mx = fmaxf(mx, vals[v]);
    }
    float sum = 0.f;
    for (int v = 0; v < 16; ++v) { vals[v] = expf(vals[v] - mx); sum += vals[v]; }
    const float inv = 1.0f / sum;
    for (int v = 0; v < 16; ++v) soft[((size_t)b * 127 + j) * 16 + v] = vals[v] * inv;
}

// ---------------------------------------------------------------------------
// es[b,vo,s] = sigmoid(tanh(...))   grid = B*127, block = 32 (16 active)
// ---------------------------------------------------------------------------
__global__ void e_es_kernel(const float* __restrict__ h_t, const float* __restrict__ h_i,
                            const float* __restrict__ cc1, const float* __restrict__ cc2,
                            const float* __restrict__ soft,
                            const float* __restrict__ Wa, const float* __restrict__ ba,
                            const float* __restrict__ Wb, const float* __restrict__ bb,
                            const float* __restrict__ Wc, const float* __restrict__ bc,
                            const float* __restrict__ Wd, const float* __restrict__ bd,
                            float* __restrict__ es)
{
    const int b = blockIdx.x / 127, s = blockIdx.x % 127, vo = threadIdx.x;
    if (vo >= 16) return;
    float a = ba[vo] + bb[vo] + bc[vo] + bd[vo];
    const float* htb = h_t + b * 128;
    const float* hib = h_i + ((size_t)b * 127 + s) * 128;
    const float* c1  = cc1 + b * 128;
    const float* c2  = cc2 + ((size_t)b * 127 + s) * 128;
    for (int k = 0; k < 128; ++k) {
        a += htb[k] * Wa[vo * 128 + k];
        a += hib[k] * Wb[vo * 128 + k];
        a += (c1[k] + c2[k]) * Wc[vo * 128 + k];
    }
    const float* sf = soft + ((size_t)b * 127 + s) * 16;
    for (int u = 0; u < 16; ++u) a += sf[u] * Wd[vo * 16 + u];
    const float e = tanhf(a);
    es[((size_t)b * 16 + vo) * 127 + s] = 1.0f / (1.0f + expf(-e));
}

// ---------------------------------------------------------------------------
// beta/ehat normalization + joint softmax over (V,T-1).  grid = B, block = 256
// ---------------------------------------------------------------------------
__global__ void attn_kernel(const float* __restrict__ es, float* __restrict__ attn)
{
    __shared__ float sE[2032];
    __shared__ float cmn[127], cmx[127], rmn[16], rmx[16];
    __shared__ float red[256];
    __shared__ float bcast[2];
    const int b = blockIdx.x, tid = threadIdx.x;

    for (int i = tid; i < 2032; i += 256) sE[i] = es[(size_t)b * 2032 + i];
    __syncthreads();

    if (tid < 127) {
        float mn = 1e30f, mx = -1e30f;
        for (int v = 0; v < 16; ++v) {
            float x = sE[v * 127 + tid];
            mn = fminf(mn, x); mx = fmaxf(mx, x);
        }
        cmn[tid] = mn; cmx[tid] = mx;
    }
    if (tid < 16) {
        float mn = 1e30f, mx = -1e30f;
        for (int t = 0; t < 127; ++t) {
            float x = sE[tid * 127 + t];
            mn = fminf(mn, x); mx = fmaxf(mx, x);
        }
        rmn[tid] = mn; rmx[tid] = mx;
    }
    __syncthreads();

    float lmx = -1e30f;
    for (int i = tid; i < 2032; i += 256) {
        const int v = i / 127, t = i % 127;
        const float x = sE[i];
        const float beta = (x - cmn[t]) / (cmx[t] - cmn[t] + 1e-5f);
        const float eh   = (x - rmn[v]) / (rmx[v] - rmn[v] + 1e-5f);
        const float val = eh * beta * 2.0f;
        sE[i] = val;
        lmx = fmaxf(lmx, val);
    }
    red[tid] = lmx; __syncthreads();
    for (int o = 128; o > 0; o >>= 1) { if (tid < o) red[tid] = fmaxf(red[tid], red[tid + o]); __syncthreads(); }
    if (tid == 0) bcast[0] = red[0];
    __syncthreads();
    const float mx = bcast[0];

    float ls = 0.f;
    for (int i = tid; i < 2032; i += 256) ls += expf(sE[i] - mx);
    red[tid] = ls; __syncthreads();
    for (int o = 128; o > 0; o >>= 1) { if (tid < o) red[tid] += red[tid + o]; __syncthreads(); }
    if (tid == 0) bcast[1] = red[0];
    __syncthreads();
    const float inv = 1.0f / bcast[1];

    for (int i = tid; i < 2032; i += 256)
        attn[(size_t)b * 2032 + i] = expf(sE[i] - mx) * inv;
}

// ---------------------------------------------------------------------------
// ctx[b,v,d] = sum_t attn[b,v,t] * h_prev[b,v,t,d]   grid = B*V, block = 256
// ---------------------------------------------------------------------------
__global__ void ctx_kernel(const float* __restrict__ attn, const float* __restrict__ hid,
                           float* __restrict__ ctx)
{
    const int bv = blockIdx.x, tid = threadIdx.x;
    float a0 = 0.f, a1 = 0.f;
    for (int t = 0; t < 127; ++t) {
        const float w = attn[bv * 127 + t];
        const float* hp = hid + ((size_t)bv * 128 + t) * 512;
        a0 += w * hp[tid];
        a1 += w * hp[tid + 256];
    }
    ctx[(size_t)bv * 512 + tid]       = a0;
    ctx[(size_t)bv * 512 + tid + 256] = a1;
}

// ---------------------------------------------------------------------------
// conv1: cat(B,2,16,512) * Wf1(16,2,5,5) pad2 -> y1(B,16,16,512)
// grid = B*16 (b,o), block = 256 (two j halves)
// ---------------------------------------------------------------------------
__global__ void conv1_kernel(const float* __restrict__ hid, const float* __restrict__ ctx,
                             const float* __restrict__ Wf1, const float* __restrict__ bf1,
                             float* __restrict__ y1)
{
    const int bo = blockIdx.x;
    const int b = bo >> 4, o = bo & 15;
    const int tid = threadIdx.x;
    for (int i = 0; i < 16; ++i) {
        for (int half = 0; half < 2; ++half) {
            const int j = tid + half * 256;
            float acc = bf1[o];
            for (int c = 0; c < 2; ++c)
                for (int dy = 0; dy < 5; ++dy) {
                    const int ii = i + dy - 2;
                    if (ii < 0 || ii > 15) continue;
                    const float* row = (c == 0)
                        ? (hid + (((size_t)b * 16 + ii) * 128 + 127) * 512)
                        : (ctx + ((size_t)b * 16 + ii) * 512);
                    const float* wk = Wf1 + ((o * 2 + c) * 5 + dy) * 5;
                    for (int dx = 0; dx < 5; ++dx) {
                        const int jj = j + dx - 2;
                        if (jj >= 0 && jj < 512) acc += row[jj] * wk[dx];
                    }
                }
            y1[(((size_t)b * 16 + o) * 16 + i) * 512 + j] = acc;
        }
    }
}

// ---------------------------------------------------------------------------
// Batch-norm stats per channel over (B, H, W).  grid = C, block = 256
// st[c] = mean, st[C+c] = rsqrt(var+eps)
// ---------------------------------------------------------------------------
__global__ void stats_kernel(const float* __restrict__ src, float* __restrict__ st,
                             int C, int HW)
{
    __shared__ float rs[256], rq[256];
    const int c = blockIdx.x, tid = threadIdx.x;
    float s = 0.f, q = 0.f;
    const int total = 32 * HW;
    for (int idx = tid; idx < total; idx += 256) {
        const int b = idx / HW, n = idx % HW;
        const float x = src[((size_t)(b * C + c)) * HW + n];
        s += x; q += x * x;
    }
    rs[tid] = s; rq[tid] = q; __syncthreads();
    for (int o = 128; o > 0; o >>= 1) {
        if (tid < o) { rs[tid] += rs[tid + o]; rq[tid] += rq[tid + o]; }
        __syncthreads();
    }
    if (tid == 0) {
        const float m = rs[0] / (float)total;
        const float var = rq[0] / (float)total - m * m;
        st[c] = m;
        st[C + c] = rsqrtf(var + 1e-5f);
    }
}

// ---------------------------------------------------------------------------
// bn + relu + 2x2 maxpool.  grid = B*C, block = Win/2, loop over Hin/2
// ---------------------------------------------------------------------------
__global__ void bnpool_kernel(const float* __restrict__ src, const float* __restrict__ st,
                              const float* __restrict__ g, const float* __restrict__ be,
                              float* __restrict__ dst, int C, int Hin, int Win)
{
    const int bc = blockIdx.x;
    const int b = bc / C, c = bc % C;
    const int jp = threadIdx.x;
    const int Ho = Hin >> 1, Wo = Win >> 1;
    const float m = st[c], rs = st[C + c], gg = g[c], bb = be[c];
    for (int ip = 0; ip < Ho; ++ip) {
        float best = -1e30f;
        for (int dy = 0; dy < 2; ++dy)
            for (int dx = 0; dx < 2; ++dx) {
                float x = src[(((size_t)(b * C + c)) * Hin + (2 * ip + dy)) * Win + (2 * jp + dx)];
                x = (x - m) * rs * gg + bb;
                x = x > 0.f ? x : 0.f;
                best = x > best ? x : best;
            }
        dst[(((size_t)(b * C + c)) * Ho + ip) * Wo + jp] = best;
    }
}

// ---------------------------------------------------------------------------
// conv2: y1p(B,16,8,256) * Wf2(32,16,5,5) pad2 -> y2(B,32,8,256)
// grid = B*32 (b,o), block = 256 (j)
// ---------------------------------------------------------------------------
__global__ void conv2_kernel(const float* __restrict__ y1p, const float* __restrict__ Wf2,
                             const float* __restrict__ bf2, float* __restrict__ y2)
{
    const int bo = blockIdx.x;
    const int b = bo >> 5, o = bo & 31;
    const int j = threadIdx.x;
    for (int i = 0; i < 8; ++i) {
        float acc = bf2[o];
        for (int c = 0; c < 16; ++c)
            for (int dy = 0; dy < 5; ++dy) {
                const int ii = i + dy - 2;
                if (ii < 0 || ii > 7) continue;
                const float* row = y1p + (((size_t)b * 16 + c) * 8 + ii) * 256;
                const float* wk = Wf2 + ((o * 16 + c) * 5 + dy) * 5;
                for (int dx = 0; dx < 5; ++dx) {
                    const int jj = j + dx - 2;
                    if (jj >= 0 && jj < 256) acc += row[jj] * wk[dx];
                }
            }
        y2[(((size_t)b * 32 + o) * 8 + i) * 256 + j] = acc;
    }
}

// ---------------------------------------------------------------------------
// logits = y2p(B,16384) @ Waff^T + baff.  grid = B, block = 256
// ---------------------------------------------------------------------------
__global__ void affine_kernel(const float* __restrict__ y, const float* __restrict__ W,
                              const float* __restrict__ bias, float* __restrict__ out)
{
    __shared__ float red[256];
    const int b = blockIdx.x, tid = threadIdx.x;
    const float* yb = y + (size_t)b * 16384;
    for (int cls = 0; cls < 10; ++cls) {
        float p = 0.f;
        for (int i = tid; i < 16384; i += 256) p += yb[i] * W[(size_t)cls * 16384 + i];
        red[tid] = p; __syncthreads();
        for (int o = 128; o > 0; o >>= 1) { if (tid < o) red[tid] += red[tid + o]; __syncthreads(); }
        if (tid == 0) out[b * 10 + cls] = red[0] + bias[cls];
        __syncthreads();
    }
}

// ---------------------------------------------------------------------------
extern "C" void kernel_launch(void* const* d_in, const int* in_sizes, int n_in,
                              void* d_out, int out_size, void* d_ws, size_t ws_size,
                              hipStream_t stream)
{
    (void)in_sizes; (void)n_in; (void)out_size; (void)ws_size;

    const float* X      = (const float*)d_in[0];
    const float* Wih0f  = (const float*)d_in[1];
    const float* Whh0f  = (const float*)d_in[2];
    const float* b0f    = (const float*)d_in[3];
    const float* Wih0b  = (const float*)d_in[4];
    const float* Whh0b  = (const float*)d_in[5];
    const float* b0b    = (const float*)d_in[6];
    const float* Wih1f  = (const float*)d_in[7];
    const float* Whh1f  = (const float*)d_in[8];
    const float* b1f    = (const float*)d_in[9];
    const float* Wih1b  = (const float*)d_in[10];
    const float* Whh1b  = (const float*)d_in[11];
    const float* b1b    = (const float*)d_in[12];
    const float* w_sc_i = (const float*)d_in[13];
    const float* b_sc_i = (const float*)d_in[14];
    const float* w_sc_t = (const float*)d_in[15];
    const float* b_sc_t = (const float*)d_in[16];
    const float* w_cc1  = (const float*)d_in[17];
    const float* b_cc1  = (const float*)d_in[18];
    const float* w_cc2  = (const float*)d_in[19];
    const float* b_cc2  = (const float*)d_in[20];
    const float* Wa     = (const float*)d_in[21];
    const float* ba     = (const float*)d_in[22];
    const float* Wb     = (const float*)d_in[23];
    const float* bb     = (const float*)d_in[24];
    const float* Wc     = (const float*)d_in[25];
    const float* bc     = (const float*)d_in[26];
    const float* Wd     = (const float*)d_in[27];
    const float* bd     = (const float*)d_in[28];
    const float* Wf1    = (const float*)d_in[29];
    const float* bf1    = (const float*)d_in[30];
    const float* Wf2    = (const float*)d_in[31];
    const float* bf2    = (const float*)d_in[32];
    const float* g1     = (const float*)d_in[33];
    const float* be1    = (const float*)d_in[34];
    const float* g2     = (const float*)d_in[35];
    const float* be2    = (const float*)d_in[36];
    const float* Waff   = (const float*)d_in[37];
    const float* baff   = (const float*)d_in[38];

    char* ws = (char*)d_ws;
    size_t off = 0;
    auto alloc = [&](size_t bytes) -> char* {
        char* p = ws + off;
        off += (bytes + 255) & ~(size_t)255;
        return p;
    };
    u16*   whh0 = (u16*)  alloc((size_t)2 * 1024 * 256 * 2);
    u16*   whh1 = (u16*)  alloc((size_t)2 * 1024 * 256 * 2);
    u16*   wih1 = (u16*)  alloc((size_t)2 * 1024 * 512 * 2);
    u16*   in1  = (u16*)  alloc((size_t)16 * 32 * 128 * 512 * 2);
    float* hid  = (float*)alloc((size_t)32 * 16 * 128 * 512 * 4);
    float* h_t  = (float*)alloc((size_t)32 * 128 * 4);
    float* h_i  = (float*)alloc((size_t)32 * 127 * 128 * 4);
    float* cc1  = (float*)alloc((size_t)32 * 128 * 4);
    float* cc2  = (float*)alloc((size_t)32 * 127 * 128 * 4);
    float* dots = (float*)alloc((size_t)32 * 16 * 127 * 4);
    float* soft = (float*)alloc((size_t)32 * 127 * 16 * 4);
    float* es   = (float*)alloc((size_t)32 * 16 * 127 * 4);
    float* attn = (float*)alloc((size_t)32 * 16 * 127 * 4);
    float* ctx  = (float*)alloc((size_t)32 * 16 * 512 * 4);
    float* y1   = (float*)alloc((size_t)32 * 16 * 16 * 512 * 4);
    float* st1  = (float*)alloc((size_t)32 * 4);
    float* y1p  = (float*)alloc((size_t)32 * 16 * 8 * 256 * 4);
    float* y2   = (float*)alloc((size_t)32 * 32 * 8 * 256 * 4);
    float* st2  = (float*)alloc((size_t)64 * 4);
    float* y2p  = (float*)alloc((size_t)32 * 32 * 4 * 128 * 4);

    // --- weight conversion to bf16 ---
    const int nWhh = 1024 * 256, nWih1 = 1024 * 512;
    cvt_bf16_kernel<<<(nWhh + 255) / 256, 256, 0, stream>>>(Whh0f, whh0,        nWhh);
    cvt_bf16_kernel<<<(nWhh + 255) / 256, 256, 0, stream>>>(Whh0b, whh0 + nWhh, nWhh);
    cvt_bf16_kernel<<<(nWhh + 255) / 256, 256, 0, stream>>>(Whh1f, whh1,        nWhh);
    cvt_bf16_kernel<<<(nWhh + 255) / 256, 256, 0, stream>>>(Whh1b, whh1 + nWhh, nWhh);
    cvt_bf16_kernel<<<(nWih1 + 255) / 256, 256, 0, stream>>>(Wih1f, wih1,         nWih1);
    cvt_bf16_kernel<<<(nWih1 + 255) / 256, 256, 0, stream>>>(Wih1b, wih1 + nWih1, nWih1);

    // --- bi-LSTM (WMMA bf16, persistent blocks, LDS-resident state) ---
    const size_t lds0 = (size_t)2 * 128 * 256 * 2 + (size_t)128 * 256 * 4 + 128 * 4; // 256.5 KB
    const size_t lds1 = (size_t)2 * 128 * 256 * 2 + (size_t)128 * 256 * 4;           // 256 KB
    lstm_l0_kernel<<<32, 256, lds0, stream>>>(X, Wih0f, Wih0b, whh0, b0f, b0b, in1);
    lstm_l1_kernel<<<32, 256, lds1, stream>>>(in1, wih1, whh1, b1f, b1b, hid);

    // --- attention pipeline ---
    ht_cc1_kernel<<<32, 128, 0, stream>>>(hid, w_sc_t, b_sc_t, w_cc1, b_cc1, h_t, cc1);
    hi_cc2_kernel<<<32 * 127, 128, 0, stream>>>(hid, w_sc_i, b_sc_i, w_cc2, b_cc2, h_i, cc2);
    dots_kernel<<<32 * 16, 128, 0, stream>>>(hid, dots);
    soft_kernel<<<32, 128, 0, stream>>>(dots, soft);
    e_es_kernel<<<32 * 127, 32, 0, stream>>>(h_t, h_i, cc1, cc2, soft,
                                             Wa, ba, Wb, bb, Wc, bc, Wd, bd, es);
    attn_kernel<<<32, 256, 0, stream>>>(es, attn);
    ctx_kernel<<<32 * 16, 256, 0, stream>>>(attn, hid, ctx);

    // --- CNN head ---
    conv1_kernel<<<32 * 16, 256, 0, stream>>>(hid, ctx, Wf1, bf1, y1);
    stats_kernel<<<16, 256, 0, stream>>>(y1, st1, 16, 16 * 512);
    bnpool_kernel<<<32 * 16, 256, 0, stream>>>(y1, st1, g1, be1, y1p, 16, 16, 512);
    conv2_kernel<<<32 * 32, 256, 0, stream>>>(y1p, Wf2, bf2, y2);
    stats_kernel<<<32, 256, 0, stream>>>(y2, st2, 32, 8 * 256);
    bnpool_kernel<<<32 * 32, 128, 0, stream>>>(y2, st2, g2, be2, y2p, 32, 8, 256);
    affine_kernel<<<32, 256, 0, stream>>>(y2p, Waff, baff, (float*)d_out);
}
